// Batch_MGCN_Policy_40097814675907
// MI455X (gfx1250) — matrix operand order repaired and checked by reading, not compiled, not run on hardware
//
#include <hip/hip_runtime.h>
#include <hip/hip_bf16.h>

// ---------------------------------------------------------------------------
// CDNA5 (gfx1250) implementation of Batch_MGCN_Policy.
// All heavy ops are dense GEMMs routed through one templated WMMA bf16 kernel
// (v_wmma_f32_16x16x32_bf16, f32 accum). Adjacency matmuls are batched via
// grid.z with the residual add fused into the epilogue. K is always a
// multiple of 32 (node feats + first weight padded 16->32). The K loop is
// software-pipelined: next tile's global loads are issued before the WMMA
// block so memory latency overlaps matrix-core work.
// ---------------------------------------------------------------------------

typedef __attribute__((ext_vector_type(16))) __bf16         v16bf;
typedef __attribute__((ext_vector_type(16))) unsigned short v16u;
typedef __attribute__((ext_vector_type(8)))  float          v8f;

#define MT 128   // block tile M
#define KT 32    // K step per WMMA

// pack two f32 -> two bf16 (truncate) in one v_perm_b32
static __device__ __forceinline__ unsigned pack_bf16x2(float lo, float hi) {
  return __builtin_amdgcn_perm(__builtin_bit_cast(unsigned, hi),
                               __builtin_bit_cast(unsigned, lo), 0x07060302u);
}
static __device__ __forceinline__ unsigned short bf16_1(float f) {
  return (unsigned short)(__builtin_bit_cast(unsigned, f) >> 16);
}

// C[M,N] = epilogue( A[M,K] @ B[K,N] ), row-major f32 in memory.
// epilogue: +bias[col] (optional), +Res[row,col] (optional), leaky_relu (act).
// Batched via grid.z with element strides sA/sB/sR/sC. Requires K % 32 == 0,
// M % 128 == 0, N % NTILE == 0, lda/ldb multiples of 4.
template <int NTILE>
__global__ __launch_bounds__(256)
void gemm_bf16_wmma(const float* __restrict__ A, long long sA, int lda,
                    const float* __restrict__ B, long long sB, int ldb,
                    const float* __restrict__ bias,
                    const float* __restrict__ Res, long long sR,
                    float* __restrict__ C, long long sC, int ldc,
                    int M, int N, int K, int act)
{
  constexpr int WBF = NTILE / 32;           // B fragments per wave (2 or 4)
  constexpr int BI  = (32 * NTILE) / 1024;  // B staging iters (2 or 4)
  __shared__ unsigned short As[MT * KT];     // [m][k]
  __shared__ unsigned short Bs[NTILE * KT];  // [n][k] (transposed tile)

  const int t    = threadIdx.x;
  const int wid  = t >> 5;
  const int lane = t & 31;
  const int l15  = lane & 15;
  const int kh   = lane >> 4;

  const long long bz = blockIdx.z;
  A += bz * sA;
  B += bz * sB;
  C += bz * sC;
  const float* Rp = Res ? (Res + bz * sR) : nullptr;

  const int mblk = blockIdx.y * MT;
  const int nblk = blockIdx.x * NTILE;
  const int wm   = (wid & 3) * 32;            // wave M offset inside tile
  const int wn   = (wid >> 2) * (NTILE / 2);  // wave N offset inside tile

  // ---- staging coordinates are K-loop invariant: hoist pointers ----
  const float* ag[4];        // global A source per staging iter
  unsigned short* al[4];     // LDS A dest per staging iter
#pragma unroll
  for (int i = 0; i < 4; ++i) {
    const int idx = (t + i * 256) * 4;          // 0..4092
    const int m = idx >> 5, k = idx & 31;
    ag[i] = A + (long long)(mblk + m) * lda + k;
    al[i] = &As[m * KT + k];
  }
  const float* bg[BI];
  unsigned short* bl[BI];
#pragma unroll
  for (int i = 0; i < BI; ++i) {
    const int idx = (t + i * 256) * 4;
    const int k = idx / NTILE, n = idx % NTILE;
    bg[i] = B + (long long)k * ldb + (nblk + n);
    bl[i] = &Bs[n * KT + k];
  }

  v8f acc[2][WBF] = {};

  // ---- software pipeline: preload tile k0=0 into registers ----
  float4 av[4], bv[BI];
#pragma unroll
  for (int i = 0; i < 4; ++i) av[i] = *(const float4*)ag[i];
#pragma unroll
  for (int i = 0; i < BI; ++i) bv[i] = *(const float4*)bg[i];

  for (int k0 = 0; k0 < K; k0 += KT) {
    // ---- convert + store current tile to LDS ----
#pragma unroll
    for (int i = 0; i < 4; ++i) {
      uint2 p;
      p.x = pack_bf16x2(av[i].x, av[i].y);
      p.y = pack_bf16x2(av[i].z, av[i].w);
      *(uint2*)al[i] = p;
    }
#pragma unroll
    for (int i = 0; i < BI; ++i) {
      bl[i][0 * KT] = bf16_1(bv[i].x);
      bl[i][1 * KT] = bf16_1(bv[i].y);
      bl[i][2 * KT] = bf16_1(bv[i].z);
      bl[i][3 * KT] = bf16_1(bv[i].w);
    }
    __syncthreads();

    // ---- issue next tile's global loads; they overlap the WMMA block ----
    const int k1 = k0 + KT;
    if (k1 < K) {
#pragma unroll
      for (int i = 0; i < 4; ++i) av[i] = *(const float4*)(ag[i] + k1);
#pragma unroll
      for (int i = 0; i < BI; ++i)
        bv[i] = *(const float4*)(bg[i] + (long long)k1 * ldb);
      // far prefetch to pre-warm L2 (gfx1250 global_prefetch_b8)
      if (k1 + KT < K) __builtin_prefetch(ag[0] + k1 + KT, 0, 1);
    }

    // ---- build fragments per ISA VGPR layouts ----
    v16bf afr[2], bfr[WBF];
#pragma unroll
    for (int f = 0; f < 2; ++f) {
      v16u au;
      const int am = wm + f * 16 + l15;
#pragma unroll
      for (int i = 0; i < 8; ++i) {
        // A 16x32 bf16: lanes0-15 VGPR0-3 K=0..7, VGPR4-7 K=16..23; lanes16-31 +8
        const int ka = (i < 4) ? (kh * 8 + 2 * i) : (16 + kh * 8 + 2 * (i - 4));
        au[2 * i]     = As[am * KT + ka];
        au[2 * i + 1] = As[am * KT + ka + 1];
      }
      afr[f] = __builtin_bit_cast(v16bf, au);
    }
#pragma unroll
    for (int f = 0; f < WBF; ++f) {
      v16u bu;
      const int bn = wn + f * 16 + l15;
#pragma unroll
      for (int i = 0; i < 8; ++i) {
        // B 32x16 bf16: lanes0-15 VGPR0-3 K=0..7, VGPR4-7 K=16..23; lanes16-31 +8
        const int kb = (i < 4) ? (kh * 8 + 2 * i) : (16 + kh * 8 + 2 * (i - 4));
        bu[2 * i]     = Bs[bn * KT + kb];
        bu[2 * i + 1] = Bs[bn * KT + kb + 1];
      }
      bfr[f] = __builtin_bit_cast(v16bf, bu);
    }

#pragma unroll
    for (int fm = 0; fm < 2; ++fm)
#pragma unroll
      for (int fn = 0; fn < WBF; ++fn)
        acc[fm][fn] = __builtin_amdgcn_wmma_f32_16x16x32_bf16(
            false, afr[fm], false, bfr[fn], (short)0, acc[fm][fn], false, false);
    __syncthreads();
  }

  // ---- epilogue: C/D layout -> lane holds col = l15, rows r + 8*kh ----
#pragma unroll
  for (int fm = 0; fm < 2; ++fm) {
#pragma unroll
    for (int fn = 0; fn < WBF; ++fn) {
      const int col = nblk + wn + fn * 16 + l15;
      const float bv2 = bias ? bias[col] : 0.0f;
#pragma unroll
      for (int r = 0; r < 8; ++r) {
        const int row = mblk + wm + fm * 16 + r + 8 * kh;
        float v = acc[fm][fn][r] + bv2;
        if (Rp) v += Rp[(long long)row * ldc + col];
        if (act) v = (v > 0.0f) ? v : 0.01f * v;
        C[(long long)row * ldc + col] = v;
      }
    }
  }
}

// pad node feats (16384 x 16) -> (16384 x 32), zero upper half
__global__ void pad_feats_kernel(const float* __restrict__ nf,
                                 float* __restrict__ out, int total) {
  int i = blockIdx.x * blockDim.x + threadIdx.x;
  if (i >= total) return;
  int row = i >> 5, c = i & 31;
  out[i] = (c < 16) ? nf[row * 16 + c] : 0.0f;
}

// pad first-layer weight (16 x 256) -> (32 x 256), zero rows 16..31
__global__ void pad_w_kernel(const float* __restrict__ W,
                             float* __restrict__ out, int total) {
  int i = blockIdx.x * blockDim.x + threadIdx.x;
  if (i >= total) return;
  int r = i >> 8;
  out[i] = (r < 16) ? W[i] : 0.0f;
}

// cat[r, 0:128]=h0[r], cat[r,128:256]=h1[r]   (16384 rows)
__global__ void concat_kernel(const float* __restrict__ h0,
                              const float* __restrict__ h1,
                              float* __restrict__ cat, int total) {
  int i = blockIdx.x * blockDim.x + threadIdx.x;
  if (i >= total) return;
  int row = i >> 8, c = i & 255;
  cat[i] = (c < 128) ? h0[row * 128 + c] : h1[row * 128 + (c - 128)];
}

// sw[b*256+r, :] = gcn_out[b*1024+r, :]  (r < 256, 64 cols)
__global__ void slice_kernel(const float* __restrict__ g,
                             float* __restrict__ sw, int total) {
  int i = blockIdx.x * blockDim.x + threadIdx.x;
  if (i >= total) return;
  int row = i >> 6, c = i & 63;
  int b = row >> 8, r = row & 255;
  sw[i] = g[((long long)(b * 1024 + r)) * 64 + c];
}

// prio[row] = X[row,:256] . W[:,0] + b   (no activation)
__global__ void dot256_kernel(const float* __restrict__ X,
                              const float* __restrict__ W,
                              const float* __restrict__ b,
                              float* __restrict__ out, int rows) {
  int row = blockIdx.x * blockDim.x + threadIdx.x;
  if (row >= rows) return;
  const float* x = X + (long long)row * 256;
  float s = b[0];
#pragma unroll 8
  for (int j = 0; j < 256; ++j) s += x[j] * W[j];
  out[row] = s;
}

// per-batch log_softmax over 256 switches; writes log_pi then probs
__global__ __launch_bounds__(256)
void logsoftmax_kernel(const float* __restrict__ prio, float* __restrict__ out) {
  __shared__ float red[256];
  const int b = blockIdx.x, t = threadIdx.x;
  const float x = prio[b * 256 + t];
  red[t] = x;
  __syncthreads();
#pragma unroll
  for (int s = 128; s > 0; s >>= 1) {
    if (t < s) red[t] = fmaxf(red[t], red[t + s]);
    __syncthreads();
  }
  const float mx = red[0];
  __syncthreads();
  red[t] = __expf(x - mx);
  __syncthreads();
#pragma unroll
  for (int s = 128; s > 0; s >>= 1) {
    if (t < s) red[t] += red[t + s];
    __syncthreads();
  }
  const float ls = x - mx - __logf(red[0]);
  out[b * 256 + t] = ls;                 // log_pi
  out[4096 + b * 256 + t] = __expf(ls);  // pi
}

// ---------------------------------------------------------------------------
// Host side
// ---------------------------------------------------------------------------
static void gemm(const float* A, long long sA, int lda,
                 const float* B, long long sB, int ldb,
                 const float* bias, const float* Res, long long sR,
                 float* C, long long sC, int ldc,
                 int M, int N, int K, int act, int batch, hipStream_t st) {
  dim3 blk(256);
  if (N % 128 == 0) {
    dim3 g(N / 128, M / MT, batch);
    hipLaunchKernelGGL(gemm_bf16_wmma<128>, g, blk, 0, st,
                       A, sA, lda, B, sB, ldb, bias, Res, sR, C, sC, ldc,
                       M, N, K, act);
  } else {
    dim3 g(N / 64, M / MT, batch);
    hipLaunchKernelGGL(gemm_bf16_wmma<64>, g, blk, 0, st,
                       A, sA, lda, B, sB, ldb, bias, Res, sR, C, sC, ldc,
                       M, N, K, act);
  }
}

extern "C" void kernel_launch(void* const* d_in, const int* in_sizes, int n_in,
                              void* d_out, int out_size, void* d_ws, size_t ws_size,
                              hipStream_t stream) {
  (void)in_sizes; (void)n_in; (void)out_size; (void)ws_size;
  const float* nf[2]  = {(const float*)d_in[0], (const float*)d_in[1]};
  const float* adj[2] = {(const float*)d_in[2], (const float*)d_in[3]};
  // params flattened in setup_inputs() insertion order, (W,b) pairs per layer
  const float* P[42];
  for (int i = 0; i < 42; ++i) P[i] = (const float*)d_in[4 + i];
  const float** featP[2] = {P + 0, P + 6};    // [F,256][256,256][256,H]
  const float** aggP[2]  = {P + 12, P + 18};  // [H,256][256,256][256,H]
  const float** mergeP   = P + 24;            // [2H,256][256,256][256,64]
  const float** fgcnP    = P + 30;            // [64,256][256,256][256,64]
  const float** prioP    = P + 36;            // [64,256][256,256][256,1]

  const int BN = 16 * 1024;       // 16384 rows
  float* ws   = (float*)d_ws;
  float* bufA = ws;                       // 16384 x 256
  float* bufB = bufA + BN * 256;          // 16384 x 256
  float* h0   = bufB + BN * 256;          // 16384 x 128
  float* h1   = h0   + BN * 128;          // 16384 x 128
  float* t128 = h1   + BN * 128;          // 16384 x 128
  float* cat  = t128 + BN * 128;          // 16384 x 256
  float* gout = cat  + BN * 256;          // 16384 x 64
  float* nfp  = gout + BN * 64;           // 16384 x 32 (padded feats)
  float* sw   = nfp  + BN * 32;           // 4096 x 64
  float* s1   = sw   + 4096 * 64;         // 4096 x 256
  float* s2   = s1   + 4096 * 256;        // 4096 x 256
  float* s3   = s2   + 4096 * 256;        // 4096 x 64
  float* pr   = s3   + 4096 * 64;         // 4096
  float* wpad = pr   + 4096;              // 32 x 256 (padded layer-1 weight)

  float* H[2] = {h0, h1};
  for (int gidx = 0; gidx < 2; ++gidx) {
    const float** fp = featP[gidx];
    // pad node feats 16 -> 32 cols and W1 16 -> 32 rows (zeros): K stays %32
    hipLaunchKernelGGL(pad_feats_kernel, dim3((BN * 32 + 255) / 256), dim3(256),
                       0, stream, nf[gidx], nfp, BN * 32);
    hipLaunchKernelGGL(pad_w_kernel, dim3((32 * 256 + 255) / 256), dim3(256),
                       0, stream, fp[0], wpad, 32 * 256);
    // feat MLP: (16384x32pad) -> 256 -> 256 -> 128, leaky_relu each layer
    gemm(nfp, 0, 32, wpad, 0, 256, fp[1], nullptr, 0, bufA, 0, 256,
         BN, 256, 32, 1, 1, stream);
    gemm(bufA, 0, 256, fp[2], 0, 256, fp[3], nullptr, 0, bufB, 0, 256,
         BN, 256, 256, 1, 1, stream);
    gemm(bufB, 0, 256, fp[4], 0, 128, fp[5], nullptr, 0, H[gidx], 0, 128,
         BN, 128, 256, 1, 1, stream);
    // 4 message-passing steps: h += adj @ agg_mlp(h)
    const float** ap = aggP[gidx];
    for (int step = 0; step < 4; ++step) {
      gemm(H[gidx], 0, 128, ap[0], 0, 256, ap[1], nullptr, 0, bufA, 0, 256,
           BN, 256, 128, 1, 1, stream);
      gemm(bufA, 0, 256, ap[2], 0, 256, ap[3], nullptr, 0, bufB, 0, 256,
           BN, 256, 256, 1, 1, stream);
      gemm(bufB, 0, 256, ap[4], 0, 128, ap[5], nullptr, 0, t128, 0, 128,
           BN, 128, 256, 1, 1, stream);
      // batched: h[b] = h[b] + adj[b](1024x1024) @ t128[b](1024x128)
      gemm(adj[gidx], 1024LL * 1024, 1024, t128, 1024LL * 128, 128,
           nullptr, H[gidx], 1024LL * 128, H[gidx], 1024LL * 128, 128,
           1024, 128, 1024, 0, 16, stream);
    }
  }

  // concat + merge MLP -> gcn_out (16384 x 64)
  {
    int total = BN * 256;
    hipLaunchKernelGGL(concat_kernel, dim3((total + 255) / 256), dim3(256), 0,
                       stream, h0, h1, cat, total);
  }
  gemm(cat, 0, 256, mergeP[0], 0, 256, mergeP[1], nullptr, 0, bufA, 0, 256,
       BN, 256, 256, 1, 1, stream);
  gemm(bufA, 0, 256, mergeP[2], 0, 256, mergeP[3], nullptr, 0, bufB, 0, 256,
       BN, 256, 256, 1, 1, stream);
  gemm(bufB, 0, 256, mergeP[4], 0, 64, mergeP[5], nullptr, 0, gout, 0, 64,
       BN, 64, 256, 1, 1, stream);

  // slice first 256 nodes per batch -> sw (4096 x 64)
  {
    int total = 4096 * 64;
    hipLaunchKernelGGL(slice_kernel, dim3((total + 255) / 256), dim3(256), 0,
                       stream, gout, sw, total);
  }
  // f_gcn_out MLP: 64 -> 256 -> 256 -> 64 (lrelu each)
  gemm(sw, 0, 64, fgcnP[0], 0, 256, fgcnP[1], nullptr, 0, s1, 0, 256,
       4096, 256, 64, 1, 1, stream);
  gemm(s1, 0, 256, fgcnP[2], 0, 256, fgcnP[3], nullptr, 0, s2, 0, 256,
       4096, 256, 256, 1, 1, stream);
  gemm(s2, 0, 256, fgcnP[4], 0, 64, fgcnP[5], nullptr, 0, s3, 0, 64,
       4096, 64, 256, 1, 1, stream);
  // prio MLP: 64 -> 256 (lrelu) -> 256 (lrelu) -> 1 (no act)
  gemm(s3, 0, 64, prioP[0], 0, 256, prioP[1], nullptr, 0, s1, 0, 256,
       4096, 256, 64, 1, 1, stream);
  gemm(s1, 0, 256, prioP[2], 0, 256, prioP[3], nullptr, 0, s2, 0, 256,
       4096, 256, 256, 1, 1, stream);
  hipLaunchKernelGGL(dot256_kernel, dim3(4096 / 256), dim3(256), 0, stream,
                     s2, prioP[4], prioP[5], pr, 4096);
  // log_softmax over switches, write (log_pi, pi)
  hipLaunchKernelGGL(logsoftmax_kernel, dim3(16), dim3(256), 0, stream,
                     pr, (float*)d_out);
}
